// SelfAttention_25855703122612
// MI455X (gfx1250) — compile-verified, matrix operand order checked by reading, not run on hardware
//
#include <hip/hip_runtime.h>

// ---------------------------------------------------------------------------
// Self-attention layer for MI455X (gfx1250): bf16 WMMA + TDM + async-to-LDS.
//   x[2,2048,1024] @ w_qkv[1024,3072] + b_qkv -> qkv  (bf16)
//   flash-attention per (b,h), S=2048, Dh=64          (bf16 WMMA + f32 softmax)
//   attn[4096,1024] @ w_o[1024,1024] + b_o -> out     (f32)
// ---------------------------------------------------------------------------

#define D_MODEL 1024
#define NHEAD   16
#define HDIM    64
#define SEQ     2048
#define BATCH   2

typedef __bf16 bf16;
typedef __attribute__((ext_vector_type(16))) __bf16 v16bf;
typedef __attribute__((ext_vector_type(8)))  __bf16 v8bf;
typedef __attribute__((ext_vector_type(8)))  float  v8f;
typedef __attribute__((ext_vector_type(4)))  unsigned int u32x4;
typedef __attribute__((ext_vector_type(8)))  int i32x8;
typedef __attribute__((ext_vector_type(4)))  int i32x4;

#if defined(__has_builtin)
# if __has_builtin(__builtin_amdgcn_tensor_load_to_lds)
#  define HAVE_TDM 1
# endif
#endif
#ifndef HAVE_TDM
# define HAVE_TDM 0
#endif

__device__ __forceinline__ v8f vzero8() { v8f z = {}; return z; }

__device__ __forceinline__ v8f wmma_bf16f32(v16bf a, v16bf b, v8f c) {
  // (neg_a, A, neg_b, B, c_mod, C, reuse_a, reuse_b)
  return __builtin_amdgcn_wmma_f32_16x16x32_bf16(false, a, false, b,
                                                 (short)0, c, false, false);
}

// Generic LDS pointer: low 32 bits hold the LDS byte offset (aperture layout).
__device__ __forceinline__ unsigned lds_off_of(const void* p) {
  return (unsigned)(unsigned long long)p;
}

// ---- CDNA5 async global->LDS copy (ASYNCcnt) ------------------------------
__device__ __forceinline__ void async_copy_b128(unsigned lds_off, const void* g) {
  asm volatile("global_load_async_to_lds_b128 %0, %1, off"
               :: "v"(lds_off), "v"(g) : "memory");
}
__device__ __forceinline__ void wait_async0() {
  asm volatile("s_wait_asynccnt 0x0" ::: "memory");
}
__device__ __forceinline__ void wait_tensor0() {
#if defined(__has_builtin) && __has_builtin(__builtin_amdgcn_s_wait_tensorcnt)
  __builtin_amdgcn_s_wait_tensorcnt(0);
#else
  asm volatile("s_wait_tensorcnt 0x0" ::: "memory");
#endif
}

// ---- CDNA5 Tensor Data Mover: one-shot 2D tile load (TENSORcnt) -----------
// D# bit layout per CDNA5 ISA; data_size = 2B.
#if HAVE_TDM
__device__ __forceinline__ void tdm_load_2d(unsigned lds_off, const void* gaddr,
                                            unsigned tensor_d0, unsigned tensor_d1,
                                            unsigned tile_d0, unsigned tile_d1,
                                            unsigned stride0_elems) {
  const unsigned long long ga = (unsigned long long)gaddr;
  u32x4 g0;
  g0[0] = 1u;                                           // count=1 (user D#)
  g0[1] = lds_off;                                      // lds_addr [63:32]
  g0[2] = (unsigned)(ga & 0xffffffffu);                 // global_addr lo
  g0[3] = (unsigned)((ga >> 32) & 0x01ffffffu)          // global_addr hi (57b)
        | (2u << 30);                                   // type=2 ("image")
  i32x8 g1;
  g1[0] = (int)(1u << 16);                              // data_size=1 -> 2 bytes
  g1[1] = (int)(tensor_d0 << 16);                       // tensor_dim0 [79:48]
  g1[2] = (int)((tensor_d0 >> 16) | (tensor_d1 << 16)); // ..dim0 hi | dim1 lo
  g1[3] = (int)((tensor_d1 >> 16) | (tile_d0 << 16));   // ..dim1 hi | tile_dim0
  g1[4] = (int)tile_d1;                                 // tile_dim1 (tile_dim2=0)
  g1[5] = (int)stride0_elems;                           // tensor_dim0_stride lo32
  g1[6] = 0;                                            // stride0 hi16 | dim1_stride lo
  g1[7] = 0;
  i32x4 z4 = {};
#if defined(__clang_major__) && (__clang_major__ >= 23)
  i32x8 z8 = {};
  __builtin_amdgcn_tensor_load_to_lds(g0, g1, z4, z4, z8, 0);
#else
  __builtin_amdgcn_tensor_load_to_lds(g0, g1, z4, z4, 0);
#endif
}
#endif

// Load one 16x32 bf16 operand fragment from LDS laid out so that the
// per-lane row (M for A, N for B) has its 32 contraction values contiguous.
// CDNA5 16-bit layout: lanes 0-15 hold K 0..7 / 16..23 in regs 0..3 / 4..7,
// lanes 16-31 hold K 8..15 / 24..31  => two contiguous ds_load_b128 per lane.
__device__ __forceinline__ v16bf load_frag(const bf16* __restrict__ p, int stride) {
  const int lane = threadIdx.x & 31;
  const bf16* r = p + (lane & 15) * stride + ((lane >> 4) << 3);
  v8bf lo = *(const v8bf*)(r);        // K = base + 0..7
  v8bf hi = *(const v8bf*)(r + 16);   // K = base + 16..23
  return __builtin_shufflevector(lo, hi, 0,1,2,3,4,5,6,7,8,9,10,11,12,13,14,15);
}

// ---------------------------------------------------------------------------
// fp32 -> bf16 conversion (one-time, bandwidth trivial)
// ---------------------------------------------------------------------------
__global__ void cvt_f32_bf16_k(const float* __restrict__ in, bf16* __restrict__ out, long n) {
  long i = (long)blockIdx.x * blockDim.x + threadIdx.x;
  const long stride = (long)gridDim.x * blockDim.x;
  for (; i < n; i += stride) out[i] = (bf16)in[i];
}

// ---------------------------------------------------------------------------
// GEMM: C[M,N] = A[M,K] * B[K,N] + bias[N]
// 128x128x32 block tile, 256 threads = 8 waves, wave tile 32x64 (2x4 WMMAs).
// A tile arrives via TDM (wave 0 issues, TENSORcnt); B tile is transposed
// into LDS by VALU (needed for the n-major fragment layout).
// ---------------------------------------------------------------------------
#define BM 128
#define BN 128
#define BK 32

template<bool STORE_BF16>
__global__ __launch_bounds__(256)
void gemm_bias_wmma(const bf16* __restrict__ A, const bf16* __restrict__ B,
                    const float* __restrict__ bias,
                    bf16* __restrict__ Cb, float* __restrict__ Cf,
                    int M, int N, int K) {
  __shared__ __align__(16) bf16 As[BM][BK];   // row-major (m, k)
  __shared__ __align__(16) bf16 Bs[BN][BK];   // n-major (n, k) == transposed

  const int tid  = threadIdx.x;
  const int wave = tid >> 5;
  const int lane = tid & 31;
  const int half = lane >> 4;
  const int wm   = wave & 3;    // 4 waves along M
  const int wn   = wave >> 2;   // 2 waves along N
  const long blockM = (long)blockIdx.y * BM;
  const long blockN = (long)blockIdx.x * BN;

  v8f acc[2][4];
#pragma unroll
  for (int i = 0; i < 2; ++i)
#pragma unroll
    for (int j = 0; j < 4; ++j) acc[i][j] = vzero8();

  // cooperative-load assignments (no bounds checks: all dims divide evenly)
  const int arow = tid >> 1;            // 128 rows, 2 threads/row (fallback path)
  const int acol = (tid & 1) * 16;
  const int brow = tid >> 3;            // 32 k-rows, 8 threads/row
  const int bcol = (tid & 7) * 16;      // 16 bf16 each
  const bf16* Ag = A + (blockM + arow) * (long)K + acol;
  const bf16* Bg = B + (long)brow * N + blockN + bcol;
  (void)Ag;

  for (int kb = 0; kb < K; kb += BK) {
    __syncthreads();
#if HAVE_TDM
    // ---- A tile: single TDM 2D-tile DMA, issued once per block by wave 0
    if (__builtin_amdgcn_readfirstlane((int)tid) < 32) {
      tdm_load_2d(lds_off_of(&As[0][0]), A + blockM * (long)K + kb,
                  (unsigned)K, (unsigned)M, BK, BM, (unsigned)K);
    }
#else
    {
      v8bf a0 = *(const v8bf*)(Ag + kb);
      v8bf a1 = *(const v8bf*)(Ag + kb + 8);
      *(v8bf*)&As[arow][acol]     = a0;
      *(v8bf*)&As[arow][acol + 8] = a1;
    }
#endif
    // ---- B tile: transpose (k,n)->(n,k) while storing to LDS
    v8bf b0 = *(const v8bf*)(Bg + (long)kb * N);
    v8bf b1 = *(const v8bf*)(Bg + (long)kb * N + 8);
#pragma unroll
    for (int e = 0; e < 8; ++e) {
      Bs[bcol + e][brow]     = b0[e];
      Bs[bcol + 8 + e][brow] = b1[e];
    }
    if (kb + BK < K) {   // emits global_prefetch_b8
      __builtin_prefetch(Bg + (long)(kb + BK) * N, 0, 1);
    }
#if HAVE_TDM
    if (__builtin_amdgcn_readfirstlane((int)tid) < 32) wait_tensor0();
#endif
    __syncthreads();

    v16bf af0 = load_frag(&As[wm * 32][0],      BK);
    v16bf af1 = load_frag(&As[wm * 32 + 16][0], BK);
#pragma unroll
    for (int j = 0; j < 4; ++j) {
      v16bf bfj = load_frag(&Bs[wn * 64 + j * 16][0], BK);
      acc[0][j] = wmma_bf16f32(af0, bfj, acc[0][j]);
      acc[1][j] = wmma_bf16f32(af1, bfj, acc[1][j]);
    }
  }

  // ---- epilogue: C layout is (M = r + 8*half, N = lane&15) per 16x16 tile
#pragma unroll
  for (int i = 0; i < 2; ++i)
#pragma unroll
    for (int j = 0; j < 4; ++j) {
      const long n = blockN + wn * 64 + j * 16 + (lane & 15);
      const float bv = bias[n];
#pragma unroll
      for (int r = 0; r < 8; ++r) {
        const long m = blockM + wm * 32 + i * 16 + half * 8 + r;
        const float v = acc[i][j][r] + bv;
        if constexpr (STORE_BF16) Cb[m * N + n] = (bf16)v;
        else                      Cf[m * N + n] = v;
      }
    }
}

// ---------------------------------------------------------------------------
// Flash attention: one block per (b, h, 128-query tile); 8 waves x 16 q-rows.
// Q tile: TDM DMA. K tile: async global->LDS (ASYNCcnt). V: VALU transpose.
// ---------------------------------------------------------------------------
#define ATT_BR 128
#define ATT_BC 64

__global__ __launch_bounds__(256)
void flash_attn_wmma(const bf16* __restrict__ qkv, bf16* __restrict__ out) {
  __shared__ __align__(16) bf16 Qs[ATT_BR][HDIM];       // 16 KB (q-row, hd)
  __shared__ __align__(16) bf16 Ks[ATT_BC][HDIM];       //  8 KB (key,  hd)  -> B-frag for S
  __shared__ __align__(16) bf16 Vs[HDIM][ATT_BC];       //  8 KB (hd,  key)  -> B-frag for O
  __shared__ __align__(16) bf16 Ps[8][16][ATT_BC];      // 16 KB wave-private P scratch

  const int b    = blockIdx.z;
  const int h    = blockIdx.y;
  const int qt   = blockIdx.x;
  const int tid  = threadIdx.x;
  const int wave = tid >> 5;
  const int lane = tid & 31;
  const int half = lane >> 4;

  const long rowStride = 3 * D_MODEL;
  const bf16* Qg = qkv + (long)b * SEQ * rowStride + h * HDIM;
  const bf16* Kg = Qg + D_MODEL;
  const bf16* Vg = Qg + 2 * D_MODEL;

  // ---- load the 128x64 Q tile once
#if HAVE_TDM
  if (__builtin_amdgcn_readfirstlane((int)tid) < 32) {
    tdm_load_2d(lds_off_of(&Qs[0][0]), Qg + (long)qt * ATT_BR * rowStride,
                (unsigned)rowStride, SEQ, HDIM, ATT_BR, (unsigned)rowStride);
    wait_tensor0();
  }
#else
  {
    const int r = tid >> 1;
    const int c = (tid & 1) * 32;
    const bf16* src = Qg + (long)(qt * ATT_BR + r) * rowStride + c;
#pragma unroll
    for (int u = 0; u < 4; ++u)
      *(v8bf*)&Qs[r][c + 8 * u] = *(const v8bf*)(src + 8 * u);
  }
#endif
  __syncthreads();

  // wave-resident Q fragments (16 rows x 64 contraction -> two 16x32 frags)
  const v16bf qa0 = load_frag(&Qs[wave * 16][0],  HDIM);
  const v16bf qa1 = load_frag(&Qs[wave * 16][32], HDIM);

  float mrow[8], lrow[8];
  v8f   oacc[4];
#pragma unroll
  for (int r = 0; r < 8; ++r) { mrow[r] = -1e30f; lrow[r] = 0.f; }
#pragma unroll
  for (int j = 0; j < 4; ++j) oacc[j] = vzero8();

  const float scale = 0.125f;   // 1/sqrt(64)

  for (int kb = 0; kb < SEQ; kb += ATT_BC) {
    __syncthreads();
    {
      const int r = tid >> 2;            // 64 rows, 4 threads/row
      const int c = (tid & 3) * 16;
      // ---- K tile: async global->LDS copy, no VGPR staging
      const bf16* ks = Kg + (long)(kb + r) * rowStride + c;
      async_copy_b128(lds_off_of(&Ks[r][c]),     ks);
      async_copy_b128(lds_off_of(&Ks[r][c + 8]), ks + 8);
      // ---- V tile: transposed (hd, key) store for the O-update B fragments
      const bf16* vsp = Vg + (long)(kb + r) * rowStride + c;
      v8bf v0 = *(const v8bf*)(vsp);
      v8bf v1 = *(const v8bf*)(vsp + 8);
#pragma unroll
      for (int e = 0; e < 8; ++e) {
        Vs[c + e][r]     = v0[e];
        Vs[c + 8 + e][r] = v1[e];
      }
    }
    wait_async0();
    __syncthreads();

    // ---- S = Q K^T : 4 column tiles x (2 wmma over hd)
    v8f sc[4];
#pragma unroll
    for (int j = 0; j < 4; ++j) {
      sc[j] = vzero8();
      v16bf kf0 = load_frag(&Ks[j * 16][0],  HDIM);
      v16bf kf1 = load_frag(&Ks[j * 16][32], HDIM);
      sc[j] = wmma_bf16f32(qa0, kf0, sc[j]);
      sc[j] = wmma_bf16f32(qa1, kf1, sc[j]);
#pragma unroll
      for (int r = 0; r < 8; ++r) sc[j][r] *= scale;
    }

    // ---- online softmax (row = r + 8*half; reduce across the 16-lane group)
#pragma unroll
    for (int r = 0; r < 8; ++r) {
      float mx = fmaxf(fmaxf(sc[0][r], sc[1][r]), fmaxf(sc[2][r], sc[3][r]));
      mx = fmaxf(mx, __shfl_xor(mx, 1, 32));
      mx = fmaxf(mx, __shfl_xor(mx, 2, 32));
      mx = fmaxf(mx, __shfl_xor(mx, 4, 32));
      mx = fmaxf(mx, __shfl_xor(mx, 8, 32));
      const float newm  = fmaxf(mrow[r], mx);
      const float alpha = __expf(mrow[r] - newm);
      mrow[r] = newm;
      float rs = 0.f;
#pragma unroll
      for (int j = 0; j < 4; ++j) {
        const float p = __expf(sc[j][r] - newm);
        sc[j][r] = p;
        rs += p;
      }
      rs += __shfl_xor(rs, 1, 32);
      rs += __shfl_xor(rs, 2, 32);
      rs += __shfl_xor(rs, 4, 32);
      rs += __shfl_xor(rs, 8, 32);
      lrow[r] = lrow[r] * alpha + rs;
#pragma unroll
      for (int j = 0; j < 4; ++j) oacc[j][r] *= alpha;
    }

    // ---- repack P: C layout -> A layout via wave-private LDS scratch
#pragma unroll
    for (int j = 0; j < 4; ++j)
#pragma unroll
      for (int r = 0; r < 8; ++r)
        Ps[wave][half * 8 + r][j * 16 + (lane & 15)] = (bf16)sc[j][r];
    // wave-private buffer: same-wave DS ops are in-order, no barrier needed
    const v16bf pa0 = load_frag(&Ps[wave][0][0],  ATT_BC);
    const v16bf pa1 = load_frag(&Ps[wave][0][32], ATT_BC);

    // ---- O += P V : 4 hd tiles x (2 wmma over keys)
#pragma unroll
    for (int j = 0; j < 4; ++j) {
      v16bf vb0 = load_frag(&Vs[j * 16][0],  ATT_BC);
      v16bf vb1 = load_frag(&Vs[j * 16][32], ATT_BC);
      oacc[j] = wmma_bf16f32(pa0, vb0, oacc[j]);
      oacc[j] = wmma_bf16f32(pa1, vb1, oacc[j]);
    }
  }

  // ---- normalize and store: out[b, s, h*64 + hd] (bf16 feed to out-proj)
#pragma unroll
  for (int r = 0; r < 8; ++r) {
    const float inv_l = 1.0f / lrow[r];
    const long  s     = (long)qt * ATT_BR + wave * 16 + half * 8 + r;
#pragma unroll
    for (int j = 0; j < 4; ++j) {
      const int hd = j * 16 + (lane & 15);
      out[((long)b * SEQ + s) * D_MODEL + h * HDIM + hd] = (bf16)(oacc[j][r] * inv_l);
    }
  }
}

// ---------------------------------------------------------------------------
// Host launcher
// ---------------------------------------------------------------------------
extern "C" void kernel_launch(void* const* d_in, const int* in_sizes, int n_in,
                              void* d_out, int out_size, void* d_ws, size_t ws_size,
                              hipStream_t stream) {
  (void)in_sizes; (void)n_in; (void)out_size; (void)ws_size;

  const float* x     = (const float*)d_in[0];   // [2,2048,1024]
  const float* w_qkv = (const float*)d_in[1];   // [1024,3072]
  const float* b_qkv = (const float*)d_in[2];   // [3072]
  const float* w_o   = (const float*)d_in[3];   // [1024,1024]
  const float* b_o   = (const float*)d_in[4];   // [1024]
  float* out = (float*)d_out;                   // [2,2048,1024]

  const long nX   = (long)BATCH * SEQ * D_MODEL;       // 4,194,304
  const long nWq  = (long)D_MODEL * 3 * D_MODEL;       // 3,145,728
  const long nWo  = (long)D_MODEL * D_MODEL;           // 1,048,576
  const long nQKV = (long)BATCH * SEQ * 3 * D_MODEL;   // 12,582,912

  bf16* xb   = (bf16*)d_ws;
  bf16* wqb  = xb  + nX;
  bf16* wob  = wqb + nWq;
  bf16* qkvb = wob + nWo;
  bf16* attb = qkvb + nQKV;                            // total ~50.3 MB

  cvt_f32_bf16_k<<<2048, 256, 0, stream>>>(x,     xb,  nX);
  cvt_f32_bf16_k<<<2048, 256, 0, stream>>>(w_qkv, wqb, nWq);
  cvt_f32_bf16_k<<<1024, 256, 0, stream>>>(w_o,   wob, nWo);

  // QKV projection: [4096,1024] x [1024,3072] + b_qkv -> bf16
  gemm_bias_wmma<true><<<dim3(3 * D_MODEL / BN, BATCH * SEQ / BM), 256, 0, stream>>>(
      xb, wqb, b_qkv, qkvb, nullptr, BATCH * SEQ, 3 * D_MODEL, D_MODEL);

  // Attention
  flash_attn_wmma<<<dim3(SEQ / ATT_BR, NHEAD, BATCH), 256, 0, stream>>>(qkvb, attb);

  // Output projection: [4096,1024] x [1024,1024] + b_o -> f32 d_out
  gemm_bias_wmma<false><<<dim3(D_MODEL / BN, BATCH * SEQ / BM), 256, 0, stream>>>(
      attb, wob, b_o, nullptr, out, BATCH * SEQ, D_MODEL, D_MODEL);
}
